// SynthesizerAttention_9165460210116
// MI455X (gfx1250) — compile-verified, matrix operand order checked by reading, not run on hardware
//
#include <hip/hip_runtime.h>
#include <hip/hip_bf16.h>

// ---------------------------------------------------------------------------
// SynthesizerAttention fused pipeline for MI455X (gfx1250, wave32, WMMA).
//   B=2, T=2048, C=1024, H=16, HD=64
// All matmuls on v_wmma_f32_16x16x32_bf16 (fp32 accumulate).
// Scores tensor (512 MB) never materialized: flash-style online softmax.
// Data layout rules:
//   - A-side tensors row-major along K (x, a, y, LDS P tile) -> 2x16B loads
//   - B-side tensors pre-transposed to [N,K] bf16 (w1, value, proj, w2)
//   - v written directly transposed [B,H,HD,T] by the producer GEMM
// GEMM K-loops use ping-pong double buffering (2x unroll, no register
// copy-back) so global_load_b128 clauses overlap the WMMAs without movs.
// Epilogues stage tiles through LDS so global stores are contiguous b128.
// ---------------------------------------------------------------------------

#define B_  2
#define T_  2048
#define C_  1024
#define H_  16
#define HD_ 64
#define BT_ (B_ * T_)
#define W2N_ 2048    // w2 is [HD, 2048]; transposed copy is [2048, HD]

typedef __bf16 bf16;
typedef __attribute__((ext_vector_type(16))) __bf16 v16bf;
typedef __attribute__((ext_vector_type(8)))  __bf16 v8bf;
typedef __attribute__((ext_vector_type(8)))  float  v8f;
typedef __attribute__((ext_vector_type(4)))  float  v4f;

// ---------------------------------------------------------------------------
// Fragment helpers (ISA 7.12.2, wave32).  Per lane the A-layout K indices are
// two contiguous runs of 8 halfwords => two 16-byte loads per fragment.
// ---------------------------------------------------------------------------
__device__ __forceinline__ v16bf load_frag8(const bf16* __restrict__ p, int ld) {
  const int lane = threadIdx.x & 31;
  const int row  = lane & 15;
  const int hh   = lane >> 4;
  const bf16* rp = p + (size_t)row * ld + hh * 8;
  const v8bf lo = *(const v8bf*)rp;          // K = hh*8 .. +7
  const v8bf hi = *(const v8bf*)(rp + 16);   // K = 16+hh*8 .. +7
  v16bf f;
#pragma unroll
  for (int e = 0; e < 8; ++e) { f[e] = lo[e]; f[8 + e] = hi[e]; }
  return f;
}

// A-fragment from the fp32 P tile staged in LDS (row-major 16x32).
__device__ __forceinline__ v16bf load_frag_p_lds(const float* __restrict__ Pl) {
  const int lane = threadIdx.x & 31;
  const int row  = lane & 15;
  const int hh   = lane >> 4;
  const float* rp = Pl + row * 32 + hh * 8;
  const v8f lo = *(const v8f*)rp;
  const v8f hi = *(const v8f*)(rp + 16);
  v16bf f;
#pragma unroll
  for (int e = 0; e < 8; ++e) { f[e] = (bf16)lo[e]; f[8 + e] = (bf16)hi[e]; }
  return f;
}

__device__ __forceinline__ v8f wmma_bf16(v16bf a, v16bf b, v8f c) {
  return __builtin_amdgcn_wmma_f32_16x16x32_bf16(
      false, a, false, b, (short)0, c, false, false);
}

// Reductions across the 16-lane half-group sharing a C/D fragment row.
__device__ __forceinline__ float hg_max(float v) {
#pragma unroll
  for (int off = 1; off < 16; off <<= 1) v = fmaxf(v, __shfl_xor(v, off, 32));
  return v;
}
__device__ __forceinline__ float hg_sum(float v) {
#pragma unroll
  for (int off = 1; off < 16; off <<= 1) v += __shfl_xor(v, off, 32);
  return v;
}

// ---------------------------------------------------------------------------
// Kernel 0a: plain fp32 -> bf16 (x stays row-major along K).
// ---------------------------------------------------------------------------
__global__ void cvt_f32_bf16(const float* __restrict__ src,
                             bf16* __restrict__ dst, int n) {
  int i = blockIdx.x * blockDim.x + threadIdx.x;
  const int stride = gridDim.x * blockDim.x;
  for (; i < n; i += stride) dst[i] = (bf16)src[i];
}

// ---------------------------------------------------------------------------
// Kernel 0b: fp32 [R,C] -> bf16 transposed [C,R], 32x32 LDS tiles (one-time;
// bf16 copies total ~6 MB and stay resident in the 192 MB L2).
// ---------------------------------------------------------------------------
__global__ void __launch_bounds__(256)
transpose_cvt(const float* __restrict__ src, bf16* __restrict__ dst,
              int rows, int cols) {
  __shared__ float tile[32][33];
  const int c0 = blockIdx.x * 32, r0 = blockIdx.y * 32;
  const int tx = threadIdx.x, ty = threadIdx.y;  // 32 x 8
#pragma unroll
  for (int j = 0; j < 32; j += 8)
    tile[ty + j][tx] = src[(size_t)(r0 + ty + j) * cols + c0 + tx];
  __syncthreads();
#pragma unroll
  for (int j = 0; j < 32; j += 8)
    dst[(size_t)(c0 + ty + j) * rows + r0 + tx] = (bf16)tile[tx][ty + j];
}

// ---------------------------------------------------------------------------
// Kernel 1: a = relu(x@w1+b1), v = x@vw+bv.  One wave = 16x64 tile of BOTH
// outputs; ping-pong K-loop (8 WMMAs per 32-K step off one x A-fragment).
//   a -> [B,H,T,HD] bf16, staged through LDS -> contiguous b128 stores
//   v -> [B,H,HD,T] bf16, packed 16B stores (D-rows are contiguous in t)
// ---------------------------------------------------------------------------
__global__ void __launch_bounds__(32)
fused_av_gemm(const bf16* __restrict__ xb,   const bf16* __restrict__ w1T,
              const bf16* __restrict__ vwT,  const float* __restrict__ w1_bias,
              const float* __restrict__ v_bias,
              bf16* __restrict__ a_out, bf16* __restrict__ vT_out) {
  __shared__ bf16 Sa[16 * 64];        // 2 KB staging for the a tile

  const int lane = threadIdx.x & 31;
  const int hh   = lane >> 4;
  const int ln   = lane & 15;
  const int m0   = blockIdx.x * 16;   // row in [BT]
  const int n0   = blockIdx.y * 64;   // col in [C]

  const bf16* xrow = xb + (size_t)m0 * C_;

  v8f accA[4] = {{}, {}, {}, {}};
  v8f accV[4] = {{}, {}, {}, {}};

  // ping-pong fragment buffers (no copy-back: 2x K-unroll alternates sets)
  v16bf afA, bwA[4], bvA[4];
  v16bf afB, bwB[4], bvB[4];

  auto load_set = [&](int k, v16bf& af, v16bf (&bw)[4], v16bf (&bv)[4]) {
    af = load_frag8(xrow + k, C_);
#pragma unroll
    for (int j = 0; j < 4; ++j) {
      bw[j] = load_frag8(w1T + (size_t)(n0 + j * 16) * C_ + k, C_);
      bv[j] = load_frag8(vwT + (size_t)(n0 + j * 16) * C_ + k, C_);
    }
  };
  auto comp_set = [&](const v16bf& af, const v16bf (&bw)[4],
                      const v16bf (&bv)[4]) {
#pragma unroll
    for (int j = 0; j < 4; ++j) {
      accA[j] = wmma_bf16(af, bw[j], accA[j]);
      accV[j] = wmma_bf16(af, bv[j], accV[j]);
    }
  };

  load_set(0, afA, bwA, bvA);
  for (int k0 = 0; k0 < C_ - 64; k0 += 64) {
    load_set(k0 + 32, afB, bwB, bvB);   // prefetch while computing A
    comp_set(afA, bwA, bvA);
    load_set(k0 + 64, afA, bwA, bvA);   // prefetch while computing B
    comp_set(afB, bwB, bvB);
  }
  load_set(C_ - 32, afB, bwB, bvB);     // peeled tail: k = C-64, C-32
  comp_set(afA, bwA, bvA);
  comp_set(afB, bwB, bvB);

  // ---- epilogue ----
  const int bb    = m0 / T_;          // 16-row tile never straddles batch
  const int t0    = m0 % T_;
  const int tbase = t0 + 8 * hh;
  const int h0    = n0 >> 6;          // n0 multiple of 64 => single head
#pragma unroll
  for (int j = 0; j < 4; ++j) {
    const int col = n0 + j * 16 + ln;
    const float ba  = w1_bias[col];
    const float bvs = v_bias[col];
    const int d = col & 63;
    // a tile -> LDS (D-layout scatter), flushed as contiguous rows below
#pragma unroll
    for (int i = 0; i < 8; ++i) {
      const float av = accA[j][i] + ba;
      Sa[(i + 8 * hh) * 64 + j * 16 + ln] = (bf16)(av > 0.f ? av : 0.f);
    }
    // v transposed [B,H,HD,T]: 8 consecutive t -> one 16B store
    v8bf vv;
#pragma unroll
    for (int i = 0; i < 8; ++i) vv[i] = (bf16)(accV[j][i] + bvs);
    *(v8bf*)(vT_out + ((size_t)(bb * H_ + h0) * HD_ + d) * T_ + tbase) = vv;
  }
  __syncthreads();
  // flush a tile: 16 rows x 64 bf16 (128 B/row); each lane moves 64 B
  {
    const bf16* s = Sa + lane * 32;
    bf16* g = a_out + (((size_t)(bb * H_ + h0)) * T_ + t0 + (lane >> 1)) * HD_ +
              (lane & 1) * 32;
#pragma unroll
    for (int q = 0; q < 4; ++q) *(v8bf*)(g + q * 8) = *(const v8bf*)(s + q * 8);
  }
}

// ---------------------------------------------------------------------------
// Kernel 2: flash-style fused  scores -> causal mask -> online softmax -> P@V.
// One wave per (b,h, 16-query tile); key blocks of 32.  V fragments are
// loaded at the top of the block so they fly during S-WMMAs + softmax.
// ---------------------------------------------------------------------------
__global__ void __launch_bounds__(32)
flash_attn(const bf16* __restrict__ a_bf, const bf16* __restrict__ vT,
           const bf16* __restrict__ w2T,  const float* __restrict__ b2,
           bf16* __restrict__ y_bf) {
  __shared__ float Pl[16 * 32];

  const int lane = threadIdx.x & 31;
  const int hh   = lane >> 4;
  const int ln   = lane & 15;
  const int bh   = blockIdx.x;           // b*H + h
  const int b    = bh / H_;
  const int h    = bh % H_;
  const int q0   = blockIdx.y * 16;

  // a-tile A-fragments (16 queries x HD=64), invariant over key loop.
  const bf16* aP = a_bf + ((size_t)bh * T_ + q0) * HD_;
  const v16bf aA0 = load_frag8(aP, HD_);        // d = 0..31
  const v16bf aA1 = load_frag8(aP + 32, HD_);   // d = 32..63

  v8f Y0 = {}, Y1 = {}, Y2 = {}, Y3 = {};
  float mrow[8], lrow[8];
#pragma unroll
  for (int i = 0; i < 8; ++i) { mrow[i] = -3e38f; lrow[i] = 0.f; }

  for (int kb = 0; kb < q0 + 16; kb += 32) {   // causal: kb <= q0+15
    // ---- issue V fragment loads early (independent of S / softmax) ----
    const bf16* vP = vT + (size_t)bh * HD_ * T_ + kb;
    const v16bf vb0 = load_frag8(vP +  0 * T_, T_);
    const v16bf vb1 = load_frag8(vP + 16 * T_, T_);
    const v16bf vb2 = load_frag8(vP + 32 * T_, T_);
    const v16bf vb3 = load_frag8(vP + 48 * T_, T_);

    // ---- S = a @ w2[:, kb:kb+32]; w2T rows are keys, ld = HD ----
    v8f S0 = {}, S1 = {};
    S0 = wmma_bf16(aA0, load_frag8(w2T + (size_t)kb * HD_,        HD_), S0);
    S0 = wmma_bf16(aA1, load_frag8(w2T + (size_t)kb * HD_ + 32,   HD_), S0);
    S1 = wmma_bf16(aA0, load_frag8(w2T + (size_t)(kb + 16) * HD_,      HD_), S1);
    S1 = wmma_bf16(aA1, load_frag8(w2T + (size_t)(kb + 16) * HD_ + 32, HD_), S1);

    const int   c0 = kb + ln,   c1 = kb + 16 + ln;
    const float bias0 = b2[c0], bias1 = b2[c1];

    // ---- bias + causal mask + online softmax ----
#pragma unroll
    for (int i = 0; i < 8; ++i) {
      const int r = i + 8 * hh;
      const int t = q0 + r;
      float s0 = S0[i] + bias0; if (c0 > t) s0 = -1e30f;
      float s1 = S1[i] + bias1; if (c1 > t) s1 = -1e30f;

      const float rm = hg_max(fmaxf(s0, s1));
      const float nm = fmaxf(mrow[i], rm);
      const float sc = __expf(mrow[i] - nm);
      mrow[i] = nm;
      const float p0 = __expf(s0 - nm);
      const float p1 = __expf(s1 - nm);
      lrow[i] = lrow[i] * sc + hg_sum(p0 + p1);
      Y0[i] *= sc; Y1[i] *= sc; Y2[i] *= sc; Y3[i] *= sc;

      Pl[r * 32 + ln]      = p0;
      Pl[r * 32 + 16 + ln] = p1;
    }
    __syncthreads();

    // ---- Y += P(16x32) @ V(32 keys x 64 hd) ----
    const v16bf pA = load_frag_p_lds(Pl);
    Y0 = wmma_bf16(pA, vb0, Y0);
    Y1 = wmma_bf16(pA, vb1, Y1);
    Y2 = wmma_bf16(pA, vb2, Y2);
    Y3 = wmma_bf16(pA, vb3, Y3);
    __syncthreads();
  }

  // ---- normalize, store y into [B,T,C] bf16 (A-side of projection) ----
#pragma unroll
  for (int i = 0; i < 8; ++i) {
    const float inv = 1.0f / lrow[i];
    const int t = q0 + i + 8 * hh;
    const size_t off = ((size_t)b * T_ + t) * C_ + h * HD_ + ln;
    y_bf[off +  0] = (bf16)(Y0[i] * inv);
    y_bf[off + 16] = (bf16)(Y1[i] * inv);
    y_bf[off + 32] = (bf16)(Y2[i] * inv);
    y_bf[off + 48] = (bf16)(Y3[i] * inv);
  }
}

// ---------------------------------------------------------------------------
// Kernel 3: out = y @ proj_w + proj_b (fp32).  16x64 tile per wave,
// ping-pong K-loop, LDS-staged contiguous output stores.
// ---------------------------------------------------------------------------
__global__ void __launch_bounds__(32)
proj_gemm(const bf16* __restrict__ yb, const bf16* __restrict__ pwT,
          const float* __restrict__ pb, float* __restrict__ out) {
  __shared__ float So[16 * 64];       // 4 KB staging for the out tile

  const int lane = threadIdx.x & 31;
  const int hh   = lane >> 4;
  const int ln   = lane & 15;
  const int m0   = blockIdx.x * 16;
  const int n0   = blockIdx.y * 64;

  const bf16* yrow = yb + (size_t)m0 * C_;

  v8f acc[4] = {{}, {}, {}, {}};
  v16bf afA, bA[4];
  v16bf afB, bB[4];

  auto load_set = [&](int k, v16bf& af, v16bf (&bf)[4]) {
    af = load_frag8(yrow + k, C_);
#pragma unroll
    for (int j = 0; j < 4; ++j)
      bf[j] = load_frag8(pwT + (size_t)(n0 + j * 16) * C_ + k, C_);
  };
  auto comp_set = [&](const v16bf& af, const v16bf (&bf)[4]) {
#pragma unroll
    for (int j = 0; j < 4; ++j) acc[j] = wmma_bf16(af, bf[j], acc[j]);
  };

  load_set(0, afA, bA);
  for (int k0 = 0; k0 < C_ - 64; k0 += 64) {
    load_set(k0 + 32, afB, bB);
    comp_set(afA, bA);
    load_set(k0 + 64, afA, bA);
    comp_set(afB, bB);
  }
  load_set(C_ - 32, afB, bB);
  comp_set(afA, bA);
  comp_set(afB, bB);

#pragma unroll
  for (int j = 0; j < 4; ++j) {
    const float bias = pb[n0 + j * 16 + ln];
#pragma unroll
    for (int i = 0; i < 8; ++i)
      So[(i + 8 * hh) * 64 + j * 16 + ln] = acc[j][i] + bias;
  }
  __syncthreads();
  // flush out tile: 16 rows x 64 f32 (256 B/row); each lane moves 128 B
  {
    const float* s = So + lane * 32;
    float* g = out + (size_t)(m0 + (lane >> 1)) * C_ + n0 + (lane & 1) * 32;
#pragma unroll
    for (int q = 0; q < 8; ++q) *(v4f*)(g + q * 4) = *(const v4f*)(s + q * 4);
  }
}

// ---------------------------------------------------------------------------
// Host-side launch. Inputs (setup_inputs order):
//   0:x 1:w1_w 2:w1_b 3:w2 4:b2 5:value_w 6:value_b 7:proj_w 8:proj_b
// ---------------------------------------------------------------------------
extern "C" void kernel_launch(void* const* d_in, const int* in_sizes, int n_in,
                              void* d_out, int out_size, void* d_ws, size_t ws_size,
                              hipStream_t stream) {
  const float* x       = (const float*)d_in[0];
  const float* w1_w    = (const float*)d_in[1];
  const float* w1_b    = (const float*)d_in[2];
  const float* w2      = (const float*)d_in[3];
  const float* b2      = (const float*)d_in[4];
  const float* value_w = (const float*)d_in[5];
  const float* value_b = (const float*)d_in[6];
  const float* proj_w  = (const float*)d_in[7];
  const float* proj_b  = (const float*)d_in[8];
  float* out = (float*)d_out;

  char* ws = (char*)d_ws;
  size_t off = 0;
  auto carve = [&](size_t bytes) -> char* {
    char* p = ws + off;
    off += (bytes + 255) & ~(size_t)255;
    return p;
  };
  bf16* x_bf  = (bf16*)carve((size_t)BT_ * C_ * 2);   // [BT, C]
  bf16* w1T   = (bf16*)carve((size_t)C_ * C_ * 2);    // [N=C, K=C]
  bf16* vwT   = (bf16*)carve((size_t)C_ * C_ * 2);    // [N=C, K=C]
  bf16* pwT   = (bf16*)carve((size_t)C_ * C_ * 2);    // [N=C, K=C]
  bf16* w2T   = (bf16*)carve((size_t)W2N_ * HD_ * 2); // [key=2048, d=64]
  bf16* a_bf  = (bf16*)carve((size_t)BT_ * C_ * 2);   // [B,H,T,HD]
  bf16* vT    = (bf16*)carve((size_t)BT_ * C_ * 2);   // [B,H,HD,T]
  bf16* y_bf  = (bf16*)carve((size_t)BT_ * C_ * 2);   // [B,T,C]
  if (off > ws_size) return;

  // Stage 0: one-time conversions / transposes (bf16 copies stay in L2).
  cvt_f32_bf16<<<2048, 256, 0, stream>>>(x, x_bf, BT_ * C_);
  dim3 tb(32, 8);
  transpose_cvt<<<dim3(C_ / 32, C_ / 32), tb, 0, stream>>>(w1_w,    w1T, C_,  C_);
  transpose_cvt<<<dim3(C_ / 32, C_ / 32), tb, 0, stream>>>(value_w, vwT, C_,  C_);
  transpose_cvt<<<dim3(C_ / 32, C_ / 32), tb, 0, stream>>>(proj_w,  pwT, C_,  C_);
  transpose_cvt<<<dim3(W2N_ / 32, HD_ / 32), tb, 0, stream>>>(w2,   w2T, HD_, W2N_);

  // Stage 1: a & v projections (one x A-fragment feeds 8 WMMAs).
  fused_av_gemm<<<dim3(BT_ / 16, C_ / 64), 32, 0, stream>>>(
      x_bf, w1T, vwT, w1_b, value_b, a_bf, vT);

  // Stage 2: fused scores + causal softmax + P@V (flash style).
  flash_attn<<<dim3(B_ * H_, T_ / 16), 32, 0, stream>>>(
      a_bf, vT, w2T, b2, y_bf);

  // Stage 3: output projection.
  proj_gemm<<<dim3(BT_ / 16, C_ / 64), 32, 0, stream>>>(
      y_bf, pwT, proj_b, out);
}
